// SoftHyperedgeGeneration_37014028157501
// MI455X (gfx1250) — matrix operand order repaired and checked by reading, not compiled, not run on hardware
//
#include <hip/hip_runtime.h>

// ---------------------------------------------------------------------------
// SoftHyperedgeGeneration for MI455X (gfx1250, wave32, WMMA bf16->f32)
// logits = (X @ Wpre^T + b_pre) @ P^T / 64  ==  X @ Q^T + c, Q = P @ Wpre
// softmax over the node axis (N).
// ---------------------------------------------------------------------------

typedef __attribute__((ext_vector_type(16))) __bf16 v16bf;
typedef __attribute__((ext_vector_type(8)))  float  v8f;

#define NUM_B   16
#define NUM_N   4096
#define DIM     512
#define NUM_M   64
#define R_TOT   (NUM_M * DIM)   // 32768 rows of W_ctx
#define CTXK    (2 * DIM)       // 1024
#define INV_SCALE 0.015625f     // 1/(sqrt(64)*8)

__device__ __forceinline__ v8f wmma_bf16(v16bf a, v16bf b, v8f c) {
    // v_wmma_f32_16x16x32_bf16 (codegen-confirmed builtin, 8 args)
    return __builtin_amdgcn_wmma_f32_16x16x32_bf16(
        false, a, false, b, (short)0, c, false, false);
}

union U16bf { v16bf v; uint4 q[2]; };

// 16 contiguous bf16 (B-fragment chunk): 32B aligned
__device__ __forceinline__ v16bf ld_bf16x16(const __bf16* p) {
    U16bf u;
    u.q[0] = ((const uint4*)p)[0];
    u.q[1] = ((const uint4*)p)[1];
    return u.v;
}

// A-fragment from bf16: two 8-element chunks (16B each)
__device__ __forceinline__ v16bf ld_bf16_2x8(const __bf16* p0, const __bf16* p1) {
    U16bf u;
    u.q[0] = *(const uint4*)p0;
    u.q[1] = *(const uint4*)p1;
    return u.v;
}

// fragment from two contiguous 8-float chunks, converted f32 -> bf16
__device__ __forceinline__ v16bf cvt_f32_2x8(const float* p0, const float* p1) {
    v16bf r;
#pragma unroll
    for (int i = 0; i < 8; ++i) r[i]     = (__bf16)p0[i];
#pragma unroll
    for (int i = 0; i < 8; ++i) r[i + 8] = (__bf16)p1[i];
    return r;
}

// B-fragment chunk from 16 contiguous f32
__device__ __forceinline__ v16bf cvt_f32_x16(const float* p) {
    v16bf r;
#pragma unroll
    for (int i = 0; i < 16; ++i) r[i] = (__bf16)p[i];
    return r;
}

// ---------------------------------------------------------------------------
// 1) context pooling: partial mean/max over N (stage 1)
// grid (16, 2, 8), block 256: d = by*256+tid, n-slice = bz*512..+512
// ---------------------------------------------------------------------------
__global__ void shg_pool_partial(const float* __restrict__ X, float* __restrict__ part) {
    int b = blockIdx.x;
    int d = blockIdx.y * 256 + threadIdx.x;
    int ns = blockIdx.z;
    const float* p = X + ((size_t)b * NUM_N + (size_t)ns * 512) * DIM + d;
    float s = 0.0f, mx = -INFINITY;
    for (int n = 0; n < 512; ++n) {
        float v = p[(size_t)n * DIM];
        s += v;
        mx = fmaxf(mx, v);
    }
    size_t base = (((size_t)b * 8 + ns) * 2) * DIM + d;
    part[base] = s;
    part[base + DIM] = mx;
}

// stage 2: ctx[b][0:512]=mean, ctx[b][512:1024]=max
__global__ void shg_pool_final(const float* __restrict__ part, float* __restrict__ ctx) {
    int b = blockIdx.x;
    int d = blockIdx.y * 256 + threadIdx.x;
    float s = 0.0f, mx = -INFINITY;
    for (int ns = 0; ns < 8; ++ns) {
        size_t base = (((size_t)b * 8 + ns) * 2) * DIM + d;
        s += part[base];
        mx = fmaxf(mx, part[base + DIM]);
    }
    ctx[(size_t)b * CTXK + d] = s * (1.0f / (float)NUM_N);
    ctx[(size_t)b * CTXK + DIM + d] = mx;
}

// ---------------------------------------------------------------------------
// 2) P[b][r] = proto[r] + b_ctx[r] + sum_k ctx[b][k] * W_ctx[r][k]   (r = m*512+d)
// WMMA: A = ctx [16 x 1024], B[k][r] = W_ctx[r][k]. One wave per 16x16 tile.
// 2048 waves -> grid 256 blocks x 8 waves. Output bf16 Pbf [B][32768].
// ---------------------------------------------------------------------------
__global__ void shg_ctx_gemm(const float* __restrict__ ctx, const float* __restrict__ Wctx,
                             const float* __restrict__ proto, const float* __restrict__ bctx,
                             __bf16* __restrict__ Pbf) {
    int wid  = blockIdx.x * 8 + (threadIdx.x >> 5);
    int lane = threadIdx.x & 31;
    int lo16 = lane & 15;
    int kh   = lane >> 4;
    int r    = wid * 16 + lo16;                 // output column (W_ctx row)

    v8f acc;
    float init = proto[r] + bctx[r];            // C-init = prototype_base + bias
#pragma unroll
    for (int j = 0; j < 8; ++j) acc[j] = init;

    const float* arow = ctx + (size_t)lo16 * CTXK;      // batch row = lane&15
    const float* brow = Wctx + (size_t)r * CTXK;
    for (int kk = 0; kk < CTXK; kk += 32) {
        v16bf a = cvt_f32_2x8(arow + kk + kh * 16, arow + kk + kh * 16 + 8);
        v16bf bb = cvt_f32_x16(brow + kk + kh * 16);
        acc = wmma_bf16(a, bb, acc);
    }
#pragma unroll
    for (int j = 0; j < 8; ++j)                 // D rows = batch j + 8*kh
        Pbf[(size_t)(j + 8 * kh) * R_TOT + r] = (__bf16)acc[j];
}

// ---------------------------------------------------------------------------
// 3a) Wtbf[k][d] = (bf16) W_pre[d][k]   (512x512 transpose, tiny)
// ---------------------------------------------------------------------------
__global__ void shg_wpre_transpose(const float* __restrict__ Wpre, __bf16* __restrict__ Wtbf) {
    int id = blockIdx.x * 256 + threadIdx.x;    // 262144
    int k = id >> 9, d = id & 511;
    Wtbf[id] = (__bf16)Wpre[(size_t)d * DIM + k];
}

// ---------------------------------------------------------------------------
// 3b) Q[b][m][k] = sum_d P[b][m][d] * W_pre[d][k] = sum_d Pbf[b][m][d]*Wtbf[k][d]
// WMMA: A = Pbf[b] [64 x 512], B[d][n=k] = Wtbf[k][d] (contiguous d per column).
// grid (16, 16) x 8 waves: wave -> tile (mt = w&3, nt = w>>2), K-loop d 512/32.
// ---------------------------------------------------------------------------
__global__ void shg_fold_gemm(const __bf16* __restrict__ Pbf, const __bf16* __restrict__ Wtbf,
                              __bf16* __restrict__ Qbf) {
    int b    = blockIdx.x;
    int w    = blockIdx.y * 8 + (threadIdx.x >> 5);  // 0..127
    int mt   = w & 3;
    int nt   = w >> 2;
    int lane = threadIdx.x & 31;
    int lo16 = lane & 15;
    int kh   = lane >> 4;

    const __bf16* pa = Pbf + (size_t)b * R_TOT + (size_t)(mt * 16 + lo16) * DIM;
    const __bf16* pb = Wtbf + (size_t)(nt * 16 + lo16) * DIM;

    v8f acc;
#pragma unroll
    for (int j = 0; j < 8; ++j) acc[j] = 0.0f;

    for (int dd = 0; dd < DIM; dd += 32) {
        v16bf a = ld_bf16_2x8(pa + dd + kh * 16, pa + dd + kh * 16 + 8);
        v16bf bb = ld_bf16x16(pb + dd + kh * 16);
        acc = wmma_bf16(a, bb, acc);
    }
    __bf16* q = Qbf + (size_t)b * R_TOT;
#pragma unroll
    for (int j = 0; j < 8; ++j)
        q[(size_t)(mt * 16 + j + 8 * kh) * DIM + nt * 16 + lo16] = (__bf16)acc[j];
}

// 3c) c[b][m] = sum_d b_pre[d] * P[b][m][d]
__global__ void shg_cvec(const __bf16* __restrict__ Pbf, const float* __restrict__ bpre,
                         float* __restrict__ cvec) {
    int b = blockIdx.x;
    int t = threadIdx.x;          // 256
    int m = t >> 2, q = t & 3;
    const __bf16* p = Pbf + (size_t)b * R_TOT + (size_t)m * DIM + q * 128;
    const float* bp = bpre + q * 128;
    float s = 0.0f;
    for (int i = 0; i < 128; ++i) s += bp[i] * (float)p[i];
    __shared__ float red[256];
    red[t] = s;
    __syncthreads();
    if (q == 0) cvec[b * NUM_M + m] = red[t] + red[t + 1] + red[t + 2] + red[t + 3];
}

// ---------------------------------------------------------------------------
// 4) logits[b][n][m] = (X[b][n][:] . Q[b][m][:] + c[b][m]) / 64
// WMMA: A = X (f32 -> bf16 on the fly), B[k][n=m] = Qbf[b][m][k] (contiguous k).
// grid (16 batches, 32 row-blocks) x 8 waves; wave = 16 rows x all 64 m (4 tiles).
// ---------------------------------------------------------------------------
__global__ void shg_logits_gemm(const float* __restrict__ X, const __bf16* __restrict__ Qbf,
                                const float* __restrict__ cvec, float* __restrict__ logits) {
    int b    = blockIdx.x;
    int rb   = blockIdx.y;
    int wave = threadIdx.x >> 5;
    int lane = threadIdx.x & 31;
    int lo16 = lane & 15;
    int kh   = lane >> 4;
    int row0 = rb * 128 + wave * 16;

    const float*  xrow = X + ((size_t)b * NUM_N + row0 + lo16) * DIM;
    const __bf16* qb   = Qbf + (size_t)b * R_TOT;
    const float*  cb   = cvec + b * NUM_M;

    v8f acc[4];
#pragma unroll
    for (int mt = 0; mt < 4; ++mt) {
        float cv = cb[mt * 16 + lo16];          // C-init carries the bias term
#pragma unroll
        for (int j = 0; j < 8; ++j) acc[mt][j] = cv;
    }

    for (int kk = 0; kk < DIM; kk += 32) {
        v16bf a = cvt_f32_2x8(xrow + kk + kh * 16, xrow + kk + kh * 16 + 8);
#pragma unroll
        for (int mt = 0; mt < 4; ++mt) {
            v16bf bb = ld_bf16x16(qb + (size_t)(mt * 16 + lo16) * DIM + kk + kh * 16);
            acc[mt] = wmma_bf16(a, bb, acc[mt]);
        }
    }

    float* lb = logits + (size_t)b * NUM_N * NUM_M;
#pragma unroll
    for (int mt = 0; mt < 4; ++mt)
#pragma unroll
        for (int j = 0; j < 8; ++j)
            lb[(size_t)(row0 + j + 8 * kh) * NUM_M + mt * 16 + lo16] = acc[mt][j] * INV_SCALE;
}

// ---------------------------------------------------------------------------
// 5) softmax over N: stats pass (per-(b,m) max and 1/sum-exp), then normalize.
// block per batch; thread t -> m = t&63 (coalesced 64-float rows), 4 n-slices.
// ---------------------------------------------------------------------------
__global__ void shg_softmax_stats(const float* __restrict__ logits, float* __restrict__ stats) {
    int b = blockIdx.x;
    int t = threadIdx.x;            // 256
    int m = t & 63, s4 = t >> 6;
    const float* lb = logits + (size_t)b * NUM_N * NUM_M;

    __shared__ float red[256];
    __shared__ float fin[64];

    float mx = -INFINITY;
    for (int n = s4; n < NUM_N; n += 4) mx = fmaxf(mx, lb[(size_t)n * NUM_M + m]);
    red[t] = mx;
    __syncthreads();
    if (t < 64)
        fin[t] = fmaxf(fmaxf(red[t], red[t + 64]), fmaxf(red[t + 128], red[t + 192]));
    __syncthreads();
    float cm = fin[m];

    float sum = 0.0f;
    for (int n = s4; n < NUM_N; n += 4) sum += __expf(lb[(size_t)n * NUM_M + m] - cm);
    __syncthreads();
    red[t] = sum;
    __syncthreads();
    if (t < 64) {
        float s = red[t] + red[t + 64] + red[t + 128] + red[t + 192];
        stats[b * NUM_M + t] = fin[t];                       // column max
        stats[NUM_B * NUM_M + b * NUM_M + t] = 1.0f / s;     // reciprocal sum
    }
}

__global__ void shg_softmax_norm(const float* __restrict__ logits, const float* __restrict__ stats,
                                 float* __restrict__ out) {
    size_t id = (size_t)blockIdx.x * 256 + threadIdx.x;      // 4194304 elements
    int m = (int)(id & 63);
    int b = (int)(id >> 18);                                 // id / (4096*64)
    float cm = stats[b * NUM_M + m];
    float rs = stats[NUM_B * NUM_M + b * NUM_M + m];
    out[id] = __expf(logits[id] - cm) * rs;
}

// ---------------------------------------------------------------------------
extern "C" void kernel_launch(void* const* d_in, const int* in_sizes, int n_in,
                              void* d_out, int out_size, void* d_ws, size_t ws_size,
                              hipStream_t stream) {
    const float* X     = (const float*)d_in[0];   // [16, 4096, 512]
    const float* proto = (const float*)d_in[1];   // [64, 512]
    const float* Wctx  = (const float*)d_in[2];   // [32768, 1024]
    const float* bctx  = (const float*)d_in[3];   // [32768]
    const float* Wpre  = (const float*)d_in[4];   // [512, 512]
    const float* bpre  = (const float*)d_in[5];   // [512]
    float* out = (float*)d_out;                   // [16, 4096, 64]

    char* w = (char*)d_ws;
    float*  part   = (float*)(w + 0);             //   524288 B: [16][8][2][512]
    float*  ctx    = (float*)(w + 524288);        //    65536 B: [16][1024]
    __bf16* Pbf    = (__bf16*)(w + 589824);       //  1048576 B: [16][32768]
    __bf16* Wtbf   = (__bf16*)(w + 1638400);      //   524288 B: [512][512]
    __bf16* Qbf    = (__bf16*)(w + 2162688);      //  1048576 B: [16][32768]
    float*  cvec   = (float*)(w + 3211264);       //     4096 B: [16][64]
    float*  stats  = (float*)(w + 3215360);       //     8192 B: max + rsum
    float*  logits = (float*)(w + 3223552);       // 16777216 B: [16][4096][64]

    shg_pool_partial  <<<dim3(16, 2, 8), 256, 0, stream>>>(X, part);
    shg_pool_final    <<<dim3(16, 2),    256, 0, stream>>>(part, ctx);
    shg_wpre_transpose<<<1024,           256, 0, stream>>>(Wpre, Wtbf);
    shg_ctx_gemm      <<<256,            256, 0, stream>>>(ctx, Wctx, proto, bctx, Pbf);
    shg_fold_gemm     <<<dim3(16, 16),   256, 0, stream>>>(Pbf, Wtbf, Qbf);
    shg_cvec          <<<16,             256, 0, stream>>>(Pbf, bpre, cvec);
    shg_logits_gemm   <<<dim3(16, 32),   256, 0, stream>>>(X, Qbf, cvec, logits);
    shg_softmax_stats <<<16,             256, 0, stream>>>(logits, stats);
    shg_softmax_norm  <<<16384,          256, 0, stream>>>(logits, stats, out);
}
